// GNNEncoder_2018634629227
// MI455X (gfx1250) — matrix operand order repaired and checked by reading, not compiled, not run on hardware
//
#include <hip/hip_runtime.h>
#include <hip/hip_bf16.h>

// ---------------- problem constants (match reference) ----------------
#define NN   4096
#define EE   131072
#define FIN  165
#define FIN_PAD 192          // FIN rounded up to multiple of 32
#define HH   128
#define HEADS 4
#define HD   32
#define LAYERS 2
#define FOUT 128
#define BN_EPS 1e-5f
#define NEG_SLOPE 0.2f
#define MASK_WPR (NN / 32)   // mask words per row = 128

typedef __attribute__((ext_vector_type(16))) _Float16 v16h;
typedef __attribute__((ext_vector_type(8)))  float    v8f;

// ---------------------------------------------------------------------
// CDNA5 async global->LDS copy (ISA §15.18.3, GLOBAL_LOAD_ASYNC_TO_LDS_B128,
// tracked by ASYNCcnt).  lds_off = low 32 bits of the generic shared-memory
// pointer (LDS aperture maps addr[31:0] directly to the LDS offset).
// ---------------------------------------------------------------------
__device__ __forceinline__ void async_ld_b128(unsigned lds_off, const float* g) {
    asm volatile("global_load_async_to_lds_b128 %0, %1, off"
                 :: "v"(lds_off), "v"((unsigned long long)(uintptr_t)g)
                 : "memory");
}
__device__ __forceinline__ void wait_async0() {
    asm volatile("s_wait_asynccnt 0x0" ::: "memory");
}
__device__ __forceinline__ unsigned lds_off_of(const void* p) {
    return (unsigned)(uintptr_t)p;
}

// =====================================================================
// adjacency mask: dense bitset, 1 bit per (row, col)
// =====================================================================
__global__ void zero_mask_kernel(unsigned* __restrict__ mask) {
    int i = blockIdx.x * blockDim.x + threadIdx.x;
    if (i < NN * MASK_WPR) mask[i] = 0u;
}

__global__ void build_mask_kernel(const int* __restrict__ ei,
                                  unsigned* __restrict__ mask) {
    int e = blockIdx.x * blockDim.x + threadIdx.x;
    if (e >= EE) return;
    int r = ei[e];        // edge_index[0][e]
    int c = ei[EE + e];   // edge_index[1][e]
    atomicOr(&mask[r * MASK_WPR + (c >> 5)], 1u << (c & 31));
}

// zero-pad x [N,165] -> xp [N,192]
__global__ void pad_x_kernel(const float* __restrict__ x, float* __restrict__ xp) {
    int i = blockIdx.x * blockDim.x + threadIdx.x;
    if (i >= NN * FIN_PAD) return;
    int r = i / FIN_PAD, c = i - r * FIN_PAD;
    xp[i] = (c < FIN) ? x[(size_t)r * FIN + c] : 0.f;
}

// zero-pad W_in [165,H] -> Wp [192,H]
__global__ void pad_w_kernel(const float* __restrict__ w, float* __restrict__ wp) {
    int i = blockIdx.x * blockDim.x + threadIdx.x;
    if (i >= FIN_PAD * HH) return;
    int r = i / HH;
    wp[i] = (r < FIN) ? w[i] : 0.f;
}

// =====================================================================
// WMMA GEMM:  C[M,Nn] = A[M,K] @ B[K,Nn] + bias[Nn]
// K multiple of 32, Nn multiple of 64.  One wave per 16(M) x 64(N) strip:
// one A operand (two b128 runs per lane) feeds 4 B tiles / 4 WMMAs per
// k-step; B slab async-copied into LDS (GLOBAL_LOAD_ASYNC_TO_LDS_B128).
// =====================================================================
__global__ void gemm_bias_wmma_kernel(const float* __restrict__ A,
                                      const float* __restrict__ B,
                                      const float* __restrict__ bias,
                                      float* __restrict__ C,
                                      int M, int K, int Nn) {
    __shared__ float sB[32 * 64];          // 8 KB weight slab
    const int lane = threadIdx.x;          // 0..31
    const int half = lane >> 4;            // 0 or 1
    const int lm   = lane & 15;
    const int tM = blockIdx.x * 16;
    const int n0 = blockIdx.y * 64;

    v8f acc[4] = {{}, {}, {}, {}};
    const float* arow = A + (size_t)(tM + lm) * K;
    const unsigned dB = lds_off_of(&sB[lane * 64]);

    for (int kb = 0; kb < K; kb += 32) {
        // ---- async stage B rows kb..kb+31, cols n0..n0+63 into LDS ----
        {
            const float* src = &B[(size_t)(kb + lane) * Nn + n0];
#pragma unroll
            for (int i = 0; i < 16; ++i)           // 64 floats per lane-row
                async_ld_b128(dB + 16 * i, src + 4 * i);
        }

        // ---- A operand: ISA 16-bit A layout = two contiguous 8-float runs ----
        v16h a;
        {
            const float4* ap0 = reinterpret_cast<const float4*>(arow + kb + 8 * half);
            const float4* ap1 = reinterpret_cast<const float4*>(arow + kb + 16 + 8 * half);
            float4 a0 = ap0[0], a1 = ap0[1];   // k = 8h .. 8h+7
            float4 a2 = ap1[0], a3 = ap1[1];   // k = 16+8h .. 16+8h+7
            a[0]  = (_Float16)a0.x; a[1]  = (_Float16)a0.y;
            a[2]  = (_Float16)a0.z; a[3]  = (_Float16)a0.w;
            a[4]  = (_Float16)a1.x; a[5]  = (_Float16)a1.y;
            a[6]  = (_Float16)a1.z; a[7]  = (_Float16)a1.w;
            a[8]  = (_Float16)a2.x; a[9]  = (_Float16)a2.y;
            a[10] = (_Float16)a2.z; a[11] = (_Float16)a2.w;
            a[12] = (_Float16)a3.x; a[13] = (_Float16)a3.y;
            a[14] = (_Float16)a3.z; a[15] = (_Float16)a3.w;
        }

        wait_async0();
        __syncthreads();

        // ---- 4 B tiles from LDS, 4 WMMAs reusing one A operand ----
#pragma unroll
        for (int t = 0; t < 4; ++t) {
            v16h b;
#pragma unroll
            for (int vi = 0; vi < 8; ++vi) {
                int kk = 2 * (vi & 3) + 16 * (vi >> 2) + 8 * half;            // B(32x16) layout k-index
                b[2 * vi]     = (_Float16)sB[kk * 64 + t * 16 + lm];
                b[2 * vi + 1] = (_Float16)sB[(kk + 1) * 64 + t * 16 + lm];
            }
            acc[t] = __builtin_amdgcn_wmma_f32_16x16x32_f16(false, a, false, b,
                                                            (short)0, acc[t], false, false);
        }
        __syncthreads();
    }

#pragma unroll
    for (int t = 0; t < 4; ++t) {
#pragma unroll
        for (int r = 0; r < 8; ++r) {
            int m = tM + r + 8 * half;          // C/D layout
            int n = n0 + t * 16 + lm;
            C[(size_t)m * Nn + n] = acc[t][r] + bias[n];
        }
    }
}

// =====================================================================
// flash-style masked GAT attention for one head / one 16-row block.
// one wave per block.  agg[m, head*HD+d] = softmax(mask(lrelu(QK^T/sqrt(d)))) @ V
// K/V chunks async-copied into LDS.
// =====================================================================
__global__ void gat_attention_kernel(const float* __restrict__ q,
                                     const float* __restrict__ k,
                                     const float* __restrict__ v,
                                     const unsigned* __restrict__ mask,
                                     float* __restrict__ agg) {
    __shared__ float sK[32 * 32];   // K rows c0..c0+31, 32 feats
    __shared__ float sV[32 * 32];
    __shared__ float sP[16 * 32];   // probability tile (C-layout -> A-layout bounce)
    __shared__ unsigned sMw[16];    // one mask word per row for this 32-col chunk

    const int head = blockIdx.y;
    const int m0   = blockIdx.x * 16;
    const int lane = threadIdx.x;
    const int half = lane >> 4;
    const int lm   = lane & 15;
    const int fofs = head * HD;

    // ---- Q operand (A layout), 1/sqrt(HD) folded in; b128 loads ----
    const float qscale = 0.1767766952966369f;   // 1/sqrt(32)
    v16h qa;
    {
        const float* qrow = q + (size_t)(m0 + lm) * HH + fofs;
        const float4* qp0 = reinterpret_cast<const float4*>(qrow + 8 * half);
        const float4* qp1 = reinterpret_cast<const float4*>(qrow + 16 + 8 * half);
        float4 q0 = qp0[0], q1 = qp0[1], q2 = qp1[0], q3 = qp1[1];
        qa[0]  = (_Float16)(q0.x * qscale); qa[1]  = (_Float16)(q0.y * qscale);
        qa[2]  = (_Float16)(q0.z * qscale); qa[3]  = (_Float16)(q0.w * qscale);
        qa[4]  = (_Float16)(q1.x * qscale); qa[5]  = (_Float16)(q1.y * qscale);
        qa[6]  = (_Float16)(q1.z * qscale); qa[7]  = (_Float16)(q1.w * qscale);
        qa[8]  = (_Float16)(q2.x * qscale); qa[9]  = (_Float16)(q2.y * qscale);
        qa[10] = (_Float16)(q2.z * qscale); qa[11] = (_Float16)(q2.w * qscale);
        qa[12] = (_Float16)(q3.x * qscale); qa[13] = (_Float16)(q3.y * qscale);
        qa[14] = (_Float16)(q3.z * qscale); qa[15] = (_Float16)(q3.w * qscale);
    }

    v8f acc0 = {}, acc1 = {};
    float run_m[8], run_l[8];
#pragma unroll
    for (int r = 0; r < 8; ++r) { run_m[r] = -1e30f; run_l[r] = 0.f; }

    const unsigned dK = lds_off_of(&sK[lane * 32]);
    const unsigned dV = lds_off_of(&sV[lane * 32]);

    for (int c0 = 0; c0 < NN; c0 += 32) {
        // ---- async stage K/V rows c0..c0+31 + mask words ----
        {
            const float* kr = &k[(size_t)(c0 + lane) * HH + fofs];
            const float* vr = &v[(size_t)(c0 + lane) * HH + fofs];
#pragma unroll
            for (int i = 0; i < 8; ++i) {          // 32 floats per lane-row
                async_ld_b128(dK + 16 * i, kr + 4 * i);
                async_ld_b128(dV + 16 * i, vr + 4 * i);
            }
            if (lane < 16) sMw[lane] = mask[(size_t)(m0 + lane) * MASK_WPR + (c0 >> 5)];
        }
        wait_async0();
        __syncthreads();

        // ---- scores = Q @ K^T for 32 columns (two 16x16 WMMAs) ----
        v16h bk0, bk1;
#pragma unroll
        for (int vi = 0; vi < 8; ++vi) {
            int kk = 2 * (vi & 3) + 16 * (vi >> 2) + 8 * half;                  // B layout k-index
            bk0[2 * vi]     = (_Float16)sK[lm * 32 + kk];        // col = node c0+lm
            bk0[2 * vi + 1] = (_Float16)sK[lm * 32 + kk + 1];
            bk1[2 * vi]     = (_Float16)sK[(16 + lm) * 32 + kk]; // col = node c0+16+lm
            bk1[2 * vi + 1] = (_Float16)sK[(16 + lm) * 32 + kk + 1];
        }
        v8f z = {};
        v8f s0 = __builtin_amdgcn_wmma_f32_16x16x32_f16(false, qa, false, bk0,
                                                        (short)0, z, false, false);
        v8f s1 = __builtin_amdgcn_wmma_f32_16x16x32_f16(false, qa, false, bk1,
                                                        (short)0, z, false, false);

        // ---- leaky relu + mask + online softmax update ----
#pragma unroll
        for (int r = 0; r < 8; ++r) {
            int m = r + 8 * half;
            unsigned w = sMw[m];
            float x0 = s0[r]; x0 = (x0 < 0.f) ? NEG_SLOPE * x0 : x0;
            float x1 = s1[r]; x1 = (x1 < 0.f) ? NEG_SLOPE * x1 : x1;
            if (!((w >> lm) & 1u))        x0 = -1e30f;
            if (!((w >> (16 + lm)) & 1u)) x1 = -1e30f;

            float tm = fmaxf(x0, x1);
#pragma unroll
            for (int off = 1; off < 16; off <<= 1)
                tm = fmaxf(tm, __shfl_xor(tm, off, 16));
            float nm    = fmaxf(run_m[r], tm);
            float scale = __expf(run_m[r] - nm);
            float p0 = __expf(x0 - nm);
            float p1 = __expf(x1 - nm);
            float ps = p0 + p1;
#pragma unroll
            for (int off = 1; off < 16; off <<= 1)
                ps += __shfl_xor(ps, off, 16);
            run_l[r] = run_l[r] * scale + ps;
            run_m[r] = nm;
            acc0[r] *= scale;
            acc1[r] *= scale;
            sP[m * 32 + lm]      = p0;   // C-layout -> LDS bounce
            sP[m * 32 + 16 + lm] = p1;
        }
        __syncthreads();

        // ---- acc += P(16x32) @ V(32x32)  (two WMMAs) ----
        v16h pa, bv0, bv1;
#pragma unroll
        for (int vi = 0; vi < 8; ++vi) {
            int ka = 2 * (vi & 3) + 16 * (vi >> 2) + 8 * half;   // A layout
            pa[2 * vi]     = (_Float16)sP[lm * 32 + ka];
            pa[2 * vi + 1] = (_Float16)sP[lm * 32 + ka + 1];
            int kb = 2 * (vi & 3) + 16 * (vi >> 2) + 8 * half;                          // B layout
            bv0[2 * vi]     = (_Float16)sV[kb * 32 + lm];
            bv0[2 * vi + 1] = (_Float16)sV[(kb + 1) * 32 + lm];
            bv1[2 * vi]     = (_Float16)sV[kb * 32 + 16 + lm];
            bv1[2 * vi + 1] = (_Float16)sV[(kb + 1) * 32 + 16 + lm];
        }
        acc0 = __builtin_amdgcn_wmma_f32_16x16x32_f16(false, pa, false, bv0,
                                                      (short)0, acc0, false, false);
        acc1 = __builtin_amdgcn_wmma_f32_16x16x32_f16(false, pa, false, bv1,
                                                      (short)0, acc1, false, false);
        __syncthreads();   // protect sK/sV/sP before next chunk overwrites
    }

    // ---- normalize and store ----
#pragma unroll
    for (int r = 0; r < 8; ++r) {
        int m = m0 + r + 8 * half;
        float inv = 1.0f / run_l[r];
        agg[(size_t)m * HH + fofs + lm]      = acc0[r] * inv;
        agg[(size_t)m * HH + fofs + 16 + lm] = acc1[r] * inv;
    }
}

// =====================================================================
// BatchNorm1d batch statistics (biased variance) over N rows
// =====================================================================
__global__ void bn_stats_kernel(const float* __restrict__ o, float* __restrict__ stats) {
    __shared__ float ss[256];
    __shared__ float sq[256];
    int ch = blockIdx.x;
    float s = 0.f, qq = 0.f;
    for (int r = threadIdx.x; r < NN; r += 256) {
        float x = o[(size_t)r * HH + ch];
        s += x; qq += x * x;
    }
    ss[threadIdx.x] = s; sq[threadIdx.x] = qq;
    __syncthreads();
    for (int st = 128; st > 0; st >>= 1) {
        if (threadIdx.x < st) {
            ss[threadIdx.x] += ss[threadIdx.x + st];
            sq[threadIdx.x] += sq[threadIdx.x + st];
        }
        __syncthreads();
    }
    if (threadIdx.x == 0) {
        float mu = ss[0] / (float)NN;
        stats[ch]      = mu;
        stats[HH + ch] = sq[0] / (float)NN - mu * mu;
    }
}

// h = relu(gamma*(o-mu)*rsqrt(var+eps)+beta) + h   (residual in place)
__global__ void bn_relu_res_kernel(const float* __restrict__ o,
                                   const float* __restrict__ stats,
                                   const float* __restrict__ gamma,
                                   const float* __restrict__ beta,
                                   float* __restrict__ h) {
    int i = blockIdx.x * blockDim.x + threadIdx.x;
    if (i >= NN * HH) return;
    int ch = i & (HH - 1);
    float mu = stats[ch], var = stats[HH + ch];
    float y = gamma[ch] * (o[i] - mu) * rsqrtf(var + BN_EPS) + beta[ch];
    h[i] = fmaxf(y, 0.f) + h[i];
}

// =====================================================================
// host-side orchestration
// =====================================================================
extern "C" void kernel_launch(void* const* d_in, const int* in_sizes, int n_in,
                              void* d_out, int out_size, void* d_ws, size_t ws_size,
                              hipStream_t stream) {
    const float* x     = (const float*)d_in[0];
    const int*   ei    = (const int*)  d_in[1];
    const float* W_in  = (const float*)d_in[2];
    const float* b_in  = (const float*)d_in[3];
    const float* Wq    = (const float*)d_in[4];
    const float* bq    = (const float*)d_in[5];
    const float* Wk    = (const float*)d_in[6];
    const float* bk    = (const float*)d_in[7];
    const float* Wv    = (const float*)d_in[8];
    const float* bv    = (const float*)d_in[9];
    const float* Wo    = (const float*)d_in[10];
    const float* bo    = (const float*)d_in[11];
    const float* gamma = (const float*)d_in[12];
    const float* beta  = (const float*)d_in[13];
    const float* W_out = (const float*)d_in[14];
    const float* b_out = (const float*)d_in[15];
    float* out = (float*)d_out;

    char* ws = (char*)d_ws;
    const size_t MB = 1u << 20;
    unsigned* mask = (unsigned*)(ws);            // 2 MB bitset
    float* xp   = (float*)(ws + 2  * MB);        // [N,192] padded input (3 MB)
    float* h    = (float*)(ws + 5  * MB);        // [N,H]
    float* qb   = (float*)(ws + 7  * MB);
    float* kb   = (float*)(ws + 9  * MB);
    float* vb   = (float*)(ws + 11 * MB);
    float* aggb = (float*)(ws + 13 * MB);
    float* Wp   = (float*)(ws + 15 * MB);        // [192,H] padded W_in (96 KB)
    float* stats= (float*)(ws + 15 * MB + 256 * 1024);
    float* ob   = qb;                            // q dead after attention -> reuse

    dim3 wave(32);

    // 1. adjacency mask + padded operands
    zero_mask_kernel<<<(NN * MASK_WPR + 255) / 256, 256, 0, stream>>>(mask);
    build_mask_kernel<<<(EE + 255) / 256, 256, 0, stream>>>(ei, mask);
    pad_x_kernel<<<(NN * FIN_PAD + 255) / 256, 256, 0, stream>>>(x, xp);
    pad_w_kernel<<<(FIN_PAD * HH + 255) / 256, 256, 0, stream>>>(W_in, Wp);

    // 2. input projection  h = xp @ Wp + b_in   (K = 192, no guards)
    gemm_bias_wmma_kernel<<<dim3(NN / 16, HH / 64), wave, 0, stream>>>(
        xp, Wp, b_in, h, NN, FIN_PAD, HH);

    // 3. GAT layers
    for (int l = 0; l < LAYERS; ++l) {
        const size_t wofs = (size_t)l * HH * HH;
        const size_t bofs = (size_t)l * HH;
        gemm_bias_wmma_kernel<<<dim3(NN / 16, HH / 64), wave, 0, stream>>>(
            h, Wq + wofs, bq + bofs, qb, NN, HH, HH);
        gemm_bias_wmma_kernel<<<dim3(NN / 16, HH / 64), wave, 0, stream>>>(
            h, Wk + wofs, bk + bofs, kb, NN, HH, HH);
        gemm_bias_wmma_kernel<<<dim3(NN / 16, HH / 64), wave, 0, stream>>>(
            h, Wv + wofs, bv + bofs, vb, NN, HH, HH);

        gat_attention_kernel<<<dim3(NN / 16, HEADS), wave, 0, stream>>>(
            qb, kb, vb, mask, aggb);

        gemm_bias_wmma_kernel<<<dim3(NN / 16, HH / 64), wave, 0, stream>>>(
            aggb, Wo + wofs, bo + bofs, ob, NN, HH, HH);

        bn_stats_kernel<<<HH, 256, 0, stream>>>(ob, stats);
        bn_relu_res_kernel<<<(NN * HH + 255) / 256, 256, 0, stream>>>(
            ob, stats, gamma + bofs, beta + bofs, h);
    }

    // 4. output projection
    gemm_bias_wmma_kernel<<<dim3(NN / 16, FOUT / 64), wave, 0, stream>>>(
        h, W_out, b_out, out, NN, HH, FOUT);
}
